// PatchCore_28501402976402
// MI455X (gfx1250) — compile-verified, matrix operand order checked by reading, not run on hardware
//
#include <hip/hip_runtime.h>

// ---------------------------------------------------------------------------
// PatchCore scoring on MI455X (gfx1250, wave32).
//   scores[i] = mean of 9 smallest euclidean distances from feat[i] to bank.
// Compute-bound: 3.3 TFLOP GEMM (~660us at ~5 PF dense bf16) vs 0.5 GB
// traffic (~21us at 23.3 TB/s). Single-pass bf16 WMMA GEMM; rank candidates
// by (bank_sq[j] - 2*dot); sqrt/mean only on the 9 survivors per row.
// B tiles staged into double-buffered LDS with GLOBAL_LOAD_ASYNC_TO_LDS_B128
// (ASYNCcnt) so the copy overlaps the WMMA phase; one barrier per K-step.
// K-loop kept at unroll 1 and no source-level fragment pipelining: the
// scheduler already places ds_load pairs optimally, and both x2 unrolling
// and register double-buffering were shown to add v_mov/v_cndmask overhead.
// ---------------------------------------------------------------------------

#define N_FEAT 16384
#define M_BANK 65536
#define DIM    1536
#define KNN    9

#define ROWS_PB 128            // rows per workgroup (8 waves x 16)
#define COLS_PC 128            // cols per chunk (8 WMMA tiles of 16)
#define CHUNKS  8              // chunks per slice
#define COLS_PS (COLS_PC * CHUNKS)   // 1024 cols per workgroup job
#define NSLICES (M_BANK / COLS_PS)   // 64
#define KSTEP  32
#define KITERS (DIM / KSTEP)         // 48

#define FLT_BIG 3.402823466e38f

typedef __bf16 bf16;
typedef __attribute__((ext_vector_type(16))) __bf16 v16bf;
typedef __attribute__((ext_vector_type(8)))  __bf16 v8bf;
typedef __attribute__((ext_vector_type(4)))  __bf16 v4bf;
typedef __attribute__((ext_vector_type(8)))  float  v8f;

// sorted ascending; t[KNN-1] is the current 9th-smallest (rejection threshold)
__device__ __forceinline__ void topk_insert(float (&t)[KNN], float c) {
  if (c < t[KNN - 1]) {
    t[KNN - 1] = c;
#pragma unroll
    for (int i = KNN - 1; i > 0; --i) {
      if (t[i] < t[i - 1]) { float tmp = t[i - 1]; t[i - 1] = t[i]; t[i] = tmp; }
    }
  }
}

// 32B global -> LDS async copy (two b128s), tracked by ASYNCcnt.
// GV mode: per-lane LDS byte address VGPR + 64-bit global address VGPR pair.
// inst offset applies to BOTH lds and memory addresses (ISA 08 §4.4).
__device__ __forceinline__ void async_stage32(unsigned lds_addr, const void* gaddr) {
  asm volatile("global_load_async_to_lds_b128 %0, %1, off\n\t"
               "global_load_async_to_lds_b128 %0, %1, off offset:16"
               :: "v"(lds_addr), "v"(gaddr) : "memory");
}
__device__ __forceinline__ void wait_async0() {
  asm volatile("s_wait_asynccnt 0x0" ::: "memory");
}

// ---------------------------------------------------------------------------
// Kernel 1: fp32 rows -> bf16 rows + squared norm per row. One wave per row.
// ---------------------------------------------------------------------------
__launch_bounds__(256)
__global__ void convert_bf16_sq(const float* __restrict__ src,
                                unsigned short* __restrict__ dstb_u,
                                float* __restrict__ sq, int rows) {
  bf16* dstb = (bf16*)dstb_u;
  const int lane = threadIdx.x & 31;
  const int w = threadIdx.x >> 5;
  const int row = blockIdx.x * 8 + w;
  if (row >= rows) return;
  const float* s = src + (size_t)row * DIM;
  bf16* d = dstb + (size_t)row * DIM;
  float acc = 0.f;
#pragma unroll
  for (int i = 0; i < DIM / 128; ++i) {
    const int k = i * 128 + lane * 4;
    float4 v = *(const float4*)(s + k);
    acc += v.x * v.x + v.y * v.y + v.z * v.z + v.w * v.w;
    v4bf o;
    o.x = (bf16)v.x; o.y = (bf16)v.y; o.z = (bf16)v.z; o.w = (bf16)v.w;
    *(v4bf*)(d + k) = o;
  }
#pragma unroll
  for (int off = 16; off > 0; off >>= 1) acc += __shfl_xor(acc, off, 32);
  if (lane == 0) sq[row] = acc;
}

// ---------------------------------------------------------------------------
// Kernel 2: bf16 WMMA GEMM + per-row running top-9 over a 1024-col slice.
// grid = (N/128 row-blocks [fastest], 64 slices) so concurrent workgroups
// share one bank slice in L2; bank streams from HBM ~once.
// ---------------------------------------------------------------------------
__launch_bounds__(256)
__global__ void patchcore_topk_partial(const unsigned short* __restrict__ featb_u,
                                       const unsigned short* __restrict__ bankb_u,
                                       const float* __restrict__ bsq,
                                       float* __restrict__ partials) {
  const bf16* featb = (const bf16*)featb_u;
  const bf16* bankb = (const bf16*)bankb_u;

  __shared__ alignas(16) bf16 Bs[2][COLS_PC][KSTEP];  // 16 KB double-buffered B
  __shared__ float Cs[8][16][65];                     // ~33 KB per-wave transpose (padded)
  __shared__ float Ms[8][16][2 * KNN];                // 9 KB lane-pair merge

  const int tid  = threadIdx.x;
  const int lane = tid & 31;
  const int w    = tid >> 5;
  const int r0   = blockIdx.x * ROWS_PB + w * 16;     // wave's 16 rows
  const int slice_c0 = blockIdx.y * COLS_PS;

  const int hi    = lane >> 4;   // K-half select per WMMA 16-bit layout
  const int l15   = lane & 15;
  const int myrow = lane >> 1;   // top-k ownership: 2 lanes per row
  const int hf    = lane & 1;

  float t9[KNN];
#pragma unroll
  for (int i = 0; i < KNN; ++i) t9[i] = FLT_BIG;

  // this thread's staging assignment (copies 32B of the B tile per K-step)
  const int scol = tid >> 1;          // 0..127
  const int sko  = (tid & 1) * 16;    // bf16 element offset within the 32-K row
  const bf16* sptr = bankb + (size_t)(slice_c0 + scol) * DIM + sko;
  const bf16* aptr = featb + (size_t)(r0 + l15) * DIM + hi * 8;
  const unsigned ldsa0 = (unsigned)(size_t)(void*)&Bs[0][scol][sko];
  const unsigned ldsa1 = (unsigned)(size_t)(void*)&Bs[1][scol][sko];

#pragma unroll 1
  for (int chunk = 0; chunk < CHUNKS; ++chunk) {
    const int c0 = slice_c0 + chunk * COLS_PC;
    const bf16* schunk = sptr + (size_t)chunk * COLS_PC * DIM;

    v8f acc[8];
#pragma unroll
    for (int t = 0; t < 8; ++t)
#pragma unroll
      for (int e = 0; e < 8; ++e) acc[t][e] = 0.f;

    // prologue: async-stage K-step 0 into buffer 0 (overlaps prior scan phase;
    // all waves have passed the previous step's barrier, so buffer 0 is free)
    async_stage32(ldsa0, schunk);

#pragma unroll 1
    for (int kk = 0; kk < KITERS; ++kk) {
      const int cur = kk & 1;
      wait_async0();                   // own async writes for 'cur' complete
      __syncthreads();                 // everyone's complete; 'nxt' no longer read
      if (kk + 1 < KITERS) {
        async_stage32(cur ? ldsa0 : ldsa1, schunk + (kk + 1) * KSTEP);
      }

      // A fragment: lane l (0-15) row r0+l, K = [hi*8,hi*8+8) and +16
      const bf16* ap = aptr + kk * KSTEP;
      v8bf a0 = *(const v8bf*)ap;
      v8bf a1 = *(const v8bf*)(ap + 16);
      v16bf afrag = __builtin_shufflevector(a0, a1, 0, 1, 2, 3, 4, 5, 6, 7,
                                            8, 9, 10, 11, 12, 13, 14, 15);
#pragma unroll
      for (int t = 0; t < 8; ++t) {
        const bf16* bp = &Bs[cur][t * 16 + l15][hi * 8];
        v8bf b0 = *(const v8bf*)bp;
        v8bf b1 = *(const v8bf*)(bp + 16);
        v16bf bfrag = __builtin_shufflevector(b0, b1, 0, 1, 2, 3, 4, 5, 6, 7,
                                              8, 9, 10, 11, 12, 13, 14, 15);
        acc[t] = __builtin_amdgcn_wmma_f32_16x16x32_bf16(
            false, afrag, false, bfrag, (short)0, acc[t], false, false);
      }
    }

    // transpose 16x128 dot tile through per-wave LDS in two 64-col passes,
    // rank candidates (bank_sq - 2*dot); two lanes per row, 32 cols each per pass.
#pragma unroll
    for (int p = 0; p < 2; ++p) {
#pragma unroll
      for (int t = 0; t < 4; ++t)
#pragma unroll
        for (int e = 0; e < 8; ++e)
          Cs[w][hi * 8 + e][t * 16 + l15] = acc[p * 4 + t][e];
      // wave-private region: per-wave DS ordering makes stores visible
#pragma unroll
      for (int j = 0; j < 32; ++j) {
        const int cc = hf * 32 + j;
        const int col = c0 + p * 64 + cc;
        const float cand = bsq[col] - 2.0f * Cs[w][myrow][cc];
        topk_insert(t9, cand);
      }
    }
  }

  // merge the two lanes of each row (both arrays sorted ascending)
#pragma unroll
  for (int i = 0; i < KNN; ++i) Ms[w][myrow][hf * KNN + i] = t9[i];
  if (hf == 0) {
    float out9[KNN];
    int ia = 0, ib = 0;
#pragma unroll
    for (int i = 0; i < KNN; ++i) {
      const float va = (ia < KNN) ? Ms[w][myrow][ia] : FLT_BIG;
      const float vb = (ib < KNN) ? Ms[w][myrow][KNN + ib] : FLT_BIG;
      if (va <= vb) { out9[i] = va; ++ia; } else { out9[i] = vb; ++ib; }
    }
    const int row = r0 + myrow;
    const size_t base = ((size_t)row * NSLICES + blockIdx.y) * KNN;
#pragma unroll
    for (int i = 0; i < KNN; ++i) partials[base + i] = out9[i];
  }
}

// ---------------------------------------------------------------------------
// Kernel 3: merge 64 partial top-9 lists per row; sqrt + mean. Wave per row.
// ---------------------------------------------------------------------------
__launch_bounds__(256)
__global__ void patchcore_final(const float* __restrict__ partials,
                                const float* __restrict__ fsq,
                                float* __restrict__ out) {
  __shared__ float R[8][32][KNN];
  const int lane = threadIdx.x & 31;
  const int w = threadIdx.x >> 5;
  const int row = blockIdx.x * 8 + w;

  float t9[KNN];
#pragma unroll
  for (int i = 0; i < KNN; ++i) t9[i] = FLT_BIG;

  for (int s = lane; s < NSLICES; s += 32) {
    const size_t base = ((size_t)row * NSLICES + s) * KNN;
#pragma unroll
    for (int i = 0; i < KNN; ++i) topk_insert(t9, partials[base + i]);
  }
#pragma unroll
  for (int i = 0; i < KNN; ++i) R[w][lane][i] = t9[i];
  __syncthreads();
  if (lane == 0) {
    float cur[KNN];
#pragma unroll
    for (int i = 0; i < KNN; ++i) cur[i] = t9[i];
    for (int l = 1; l < 32; ++l)
#pragma unroll
      for (int i = 0; i < KNN; ++i) topk_insert(cur, R[w][l][i]);
    const float fs = fsq[row];
    float acc = 0.f;
#pragma unroll
    for (int i = 0; i < KNN; ++i) acc += sqrtf(fmaxf(fs + cur[i], 0.f));
    out[row] = acc * (1.0f / KNN);
  }
}

// ---------------------------------------------------------------------------
extern "C" void kernel_launch(void* const* d_in, const int* in_sizes, int n_in,
                              void* d_out, int out_size, void* d_ws, size_t ws_size,
                              hipStream_t stream) {
  (void)in_sizes; (void)n_in; (void)out_size; (void)ws_size;
  const float* feat = (const float*)d_in[0];
  const float* bank = (const float*)d_in[1];
  float* out = (float*)d_out;

  char* ws = (char*)d_ws;
  size_t off = 0;
  auto carve = [&](size_t bytes) -> void* {
    void* p = ws + off;
    off = (off + bytes + 255) & ~(size_t)255;
    return p;
  };
  unsigned short* bankb = (unsigned short*)carve((size_t)M_BANK * DIM * 2);  // 192 MB
  unsigned short* featb = (unsigned short*)carve((size_t)N_FEAT * DIM * 2);  //  48 MB
  float* bsq = (float*)carve((size_t)M_BANK * 4);
  float* fsq = (float*)carve((size_t)N_FEAT * 4);
  float* partials = (float*)carve((size_t)N_FEAT * NSLICES * KNN * 4);       //  36 MB

  convert_bf16_sq<<<M_BANK / 8, 256, 0, stream>>>(bank, bankb, bsq, M_BANK);
  convert_bf16_sq<<<N_FEAT / 8, 256, 0, stream>>>(feat, featb, fsq, N_FEAT);
  patchcore_topk_partial<<<dim3(N_FEAT / ROWS_PB, NSLICES), 256, 0, stream>>>(
      featb, bankb, bsq, partials);
  patchcore_final<<<N_FEAT / 8, 256, 0, stream>>>(partials, fsq, out);
}
